// Model_44212393345082
// MI455X (gfx1250) — matrix operand order repaired and checked by reading, not compiled, not run on hardware
//
#include <hip/hip_runtime.h>

typedef __attribute__((ext_vector_type(16))) __bf16 v16bf;
typedef __attribute__((ext_vector_type(8)))  float  v8f;
typedef __attribute__((ext_vector_type(4)))  int    v4i;

// ---------------------------------------------------------------------------
// WMMA helpers (CDNA5 wave32, D = A(16x32 bf16) x B(32x16 bf16) + C(16x16 f32))
// ---------------------------------------------------------------------------
static __device__ inline v8f wmma_bf16(v16bf a, v16bf b, v8f c) {
    return __builtin_amdgcn_wmma_f32_16x16x32_bf16(
        /*neg_a=*/false, a, /*neg_b=*/false, b,
        /*c_mod=*/(short)0, c, /*reuse_a=*/false, /*reuse_b=*/false);
}

// A fragment from an LDS tile stored row-major (pitch in bf16 elements).
static __device__ inline v16bf load_a_frag(const __bf16* sA, int pitch, int k0, int lane) {
    const int row = lane & 15, hi = lane >> 4;
    const __bf16* p = sA + row * pitch + k0 + hi * 8;
    union { uint4 u[2]; v16bf v; } f;
    f.u[0] = *(const uint4*)(p);
    f.u[1] = *(const uint4*)(p + 16);
    return f.v;
}

// B fragment from fragment-ordered weight buffer (8 dwords / lane, contiguous).
static __device__ inline v16bf load_b_frag(const unsigned* Wf, int lane) {
    const uint4* p = (const uint4*)(Wf + lane * 8);
    union { uint4 u[2]; v16bf v; } f;
    f.u[0] = p[0];
    f.u[1] = p[1];
    return f.v;
}

static __device__ inline void gemm_tile(const __bf16* sA, int pitch, int KT,
                                        const unsigned* Wf, int lane, v8f acc[4]) {
    for (int kt = 0; kt < KT; ++kt) {
        v16bf a = load_a_frag(sA, pitch, kt * 32, lane);
#pragma unroll
        for (int nt = 0; nt < 4; ++nt) {
            v16bf b = load_b_frag(Wf + ((kt * 4 + nt) << 8), lane);
            acc[nt] = wmma_bf16(a, b, acc[nt]);
        }
    }
}

// Two M-tiles (32 rows) sharing each B fragment: halves weight traffic.
static __device__ inline void gemm_tile2(const __bf16* sA0, const __bf16* sA1, int pitch,
                                         int KT, const unsigned* Wf, int lane,
                                         v8f acc0[4], v8f acc1[4]) {
    for (int kt = 0; kt < KT; ++kt) {
        v16bf a0 = load_a_frag(sA0, pitch, kt * 32, lane);
        v16bf a1 = load_a_frag(sA1, pitch, kt * 32, lane);
#pragma unroll
        for (int nt = 0; nt < 4; ++nt) {
            v16bf b = load_b_frag(Wf + ((kt * 4 + nt) << 8), lane);
            acc0[nt] = wmma_bf16(a0, b, acc0[nt]);
            acc1[nt] = wmma_bf16(a1, b, acc1[nt]);
        }
    }
}

// bias + ReLU + bf16 convert, C layout: row=(l>>4)*8+v, col=(l&15)+nt*16.
static __device__ inline void store_c_relu(__bf16* sH, const v8f acc[4],
                                           const float* bias, int lane) {
    const int col = lane & 15, hi = lane >> 4;
#pragma unroll
    for (int nt = 0; nt < 4; ++nt) {
        float b = bias[nt * 16 + col];
#pragma unroll
        for (int v = 0; v < 8; ++v) {
            float val = acc[nt][v] + b;
            val = fmaxf(val, 0.0f);
            sH[(hi * 8 + v) * 64 + nt * 16 + col] = (__bf16)val;
        }
    }
}

// Coalesced copy of nchunks*16B from LDS to contiguous global rows.
static __device__ inline void copy_tiles_out(__bf16* dst, const __bf16* sH, int lane, int nchunks) {
    const uint4* s = (const uint4*)sH;
    uint4* d = (uint4*)dst;
    for (int i = lane; i < nchunks; i += 32) d[i] = s[i];
}

static __device__ inline void zero_acc(v8f acc[4]) {
    v8f z = {0.f, 0.f, 0.f, 0.f, 0.f, 0.f, 0.f, 0.f};
#pragma unroll
    for (int i = 0; i < 4; ++i) acc[i] = z;
}

static __device__ inline void atomic_add_f32(float* p, float v) {
    __hip_atomic_fetch_add(p, v, __ATOMIC_RELAXED, __HIP_MEMORY_SCOPE_AGENT);
}

// ---------------------------------------------------------------------------
// Async global -> LDS (ASYNCcnt path), with inline-asm fallback.
// Builtin signature (from hipcc diagnostic): (v4i AS1*, v4i AS3*, imm, imm).
// ---------------------------------------------------------------------------
static __device__ inline void async_load_lds_b128(const void* gsrc, void* ldst) {
#if __has_builtin(__builtin_amdgcn_global_load_async_to_lds_b128)
    __builtin_amdgcn_global_load_async_to_lds_b128(
        (__attribute__((address_space(1))) v4i*)gsrc,
        (__attribute__((address_space(3))) v4i*)ldst,
        /*offset=*/0, /*cpol=*/0);
#else
    unsigned long long ga = (unsigned long long)gsrc;
    unsigned la = (unsigned)(unsigned long long)ldst;  // low 32 bits = LDS offset
    asm volatile("global_load_async_to_lds_b128 %0, %1, off"
                 :: "v"(la), "v"(ga) : "memory");
#endif
}

static __device__ inline void wait_async0() {
#if __has_builtin(__builtin_amdgcn_s_wait_asynccnt)
    __builtin_amdgcn_s_wait_asynccnt(0);
#else
    asm volatile("s_wait_asynccnt 0" ::: "memory");
#endif
}

// ---------------------------------------------------------------------------
// Weight prep: f32 [N=64][K] row-major -> bf16 fragment order
// ---------------------------------------------------------------------------
__global__ __launch_bounds__(256) void prep_weight_kernel(const float* __restrict__ W,
                                                          unsigned* __restrict__ out,
                                                          int K, int KT, int NT) {
    int tid = blockIdx.x * blockDim.x + threadIdx.x;
    int total = KT * NT * 256;
    if (tid >= total) return;
    int j = tid & 7;
    int l = (tid >> 3) & 31;
    int rest = tid >> 8;
    int nt = rest % NT;
    int kt = rest / NT;
    int n = nt * 16 + (l & 15);
    int k = kt * 32 + ((l >> 4) << 3) + ((j >> 2) << 4) + 2 * (j & 3);
    __bf16 lo = (__bf16)W[(size_t)n * K + k];
    __bf16 hi = (__bf16)W[(size_t)n * K + k + 1];
    unsigned u = ((unsigned)__builtin_bit_cast(unsigned short, hi) << 16) |
                 (unsigned)__builtin_bit_cast(unsigned short, lo);
    out[tid] = u;
}

__global__ __launch_bounds__(256) void zero_f32_kernel(float* __restrict__ p, int n) {
    int i = blockIdx.x * blockDim.x + threadIdx.x;
    if (i < n) p[i] = 0.0f;
}

// ---------------------------------------------------------------------------
// Node encoder: x[N,1] -> relu(x*w1+b1) -> relu(@w2.T+b2) -> enc_x, xx (bf16)
// ---------------------------------------------------------------------------
__global__ __launch_bounds__(128) void node_encoder_kernel(
    const float* __restrict__ x, const float* __restrict__ w1, const float* __restrict__ b1,
    const unsigned* __restrict__ w2f, const float* __restrict__ b2,
    __bf16* __restrict__ enc_x, __bf16* __restrict__ xx, int ntiles) {
    __shared__ __bf16 sH1[4][1024];
    __shared__ __bf16 sH2[4][1024];
    const int wave = threadIdx.x >> 5, lane = threadIdx.x & 31;
    const int t = blockIdx.x * 4 + wave;
    if (t >= ntiles) return;
    const int n0 = t * 16;
    for (int r = 0; r < 16; ++r) {
        float xv = x[n0 + r];
        for (int o = lane; o < 64; o += 32) {
            float h = fmaxf(xv * w1[o] + b1[o], 0.0f);
            sH1[wave][r * 64 + o] = (__bf16)h;
        }
    }
    v8f acc[4];
    zero_acc(acc);
    gemm_tile(sH1[wave], 64, 2, w2f, lane, acc);
    store_c_relu(sH2[wave], acc, b2, lane);
    copy_tiles_out(enc_x + (size_t)n0 * 64, sH2[wave], lane, 128);
    copy_tiles_out(xx + (size_t)n0 * 64, sH2[wave], lane, 128);
}

// ---------------------------------------------------------------------------
// Edge encoder: edge_attr[E,3] -> relu(@w1.T+b1) -> relu(@w2.T+b2) -> enc_e, ee
// ---------------------------------------------------------------------------
__global__ __launch_bounds__(128) void edge_encoder_kernel(
    const float* __restrict__ ea, const float* __restrict__ w1, const float* __restrict__ b1,
    const unsigned* __restrict__ w2f, const float* __restrict__ b2,
    __bf16* __restrict__ enc_e, __bf16* __restrict__ ee, int ntiles) {
    __shared__ __bf16 sH1[4][1024];
    __shared__ __bf16 sH2[4][1024];
    const int wave = threadIdx.x >> 5, lane = threadIdx.x & 31;
    const int t = blockIdx.x * 4 + wave;
    if (t >= ntiles) return;
    const int e0 = t * 16;
    for (int r = 0; r < 16; ++r) {
        const float a0 = ea[(size_t)(e0 + r) * 3 + 0];
        const float a1 = ea[(size_t)(e0 + r) * 3 + 1];
        const float a2 = ea[(size_t)(e0 + r) * 3 + 2];
        for (int o = lane; o < 64; o += 32) {
            float h = a0 * w1[o * 3] + a1 * w1[o * 3 + 1] + a2 * w1[o * 3 + 2] + b1[o];
            sH1[wave][r * 64 + o] = (__bf16)fmaxf(h, 0.0f);
        }
    }
    v8f acc[4];
    zero_acc(acc);
    gemm_tile(sH1[wave], 64, 2, w2f, lane, acc);
    store_c_relu(sH2[wave], acc, b2, lane);
    copy_tiles_out(enc_e + (size_t)e0 * 64, sH2[wave], lane, 128);
    copy_tiles_out(ee + (size_t)e0 * 64, sH2[wave], lane, 128);
}

// ---------------------------------------------------------------------------
// Edge update, M=32 per wave (two 16-row WMMA tiles share each B fragment):
//   A = [xx[row] | xx[col] | ee | enc_e]  (32x256 bf16 tile in LDS, async gather)
//   ee = relu(relu(A@W1.T+b1)@W2.T+b2);  recvF[col] += ee  (f32 atomics)
// ---------------------------------------------------------------------------
__global__ __launch_bounds__(64) void edge_update_kernel(
    const __bf16* __restrict__ xx, const __bf16* __restrict__ enc_e, __bf16* __restrict__ ee,
    const int* __restrict__ ei, int E,
    const unsigned* __restrict__ w1f, const float* __restrict__ b1,
    const unsigned* __restrict__ w2f, const float* __restrict__ b2,
    float* __restrict__ recvF, int ntiles) {
    __shared__ __bf16 sA[2][32 * 256];   // 16 KB / wave
    __shared__ __bf16 sH[2][32 * 64];    //  4 KB / wave
    const int wave = threadIdx.x >> 5, lane = threadIdx.x & 31;
    const int t = blockIdx.x * 2 + wave;
    if (t >= ntiles) return;
    const int e0 = t * 32;
    __bf16* sAw = sA[wave];
    __bf16* sHw = sH[wave];

    // Async gather straight into LDS: one 16B chunk per lane per row.
    const int seg = lane >> 3, q = lane & 7;
    for (int r = 0; r < 32; ++r) {
        const int e = e0 + r;
        const uint4* src;
        if (seg == 0)      src = (const uint4*)(xx + (size_t)ei[e] * 64) + q;
        else if (seg == 1) src = (const uint4*)(xx + (size_t)ei[E + e] * 64) + q;
        else if (seg == 2) src = (const uint4*)(ee + (size_t)e * 64) + q;
        else               src = (const uint4*)(enc_e + (size_t)e * 64) + q;
        async_load_lds_b128(src, sAw + r * 256 + (seg * 8 + q) * 8);
    }
    wait_async0();

    v8f acc0[4], acc1[4];
    zero_acc(acc0); zero_acc(acc1);
    gemm_tile2(sAw, sAw + 16 * 256, 256, 8, w1f, lane, acc0, acc1);   // 256 -> 64
    store_c_relu(sHw, acc0, b1, lane);
    store_c_relu(sHw + 16 * 64, acc1, b1, lane);

    zero_acc(acc0); zero_acc(acc1);
    gemm_tile2(sHw, sHw + 16 * 64, 64, 2, w2f, lane, acc0, acc1);     // 64 -> 64
    store_c_relu(sAw, acc0, b2, lane);             // reuse sA head as 32x64 out tile
    store_c_relu(sAw + 16 * 64, acc1, b2, lane);

    copy_tiles_out(ee + (size_t)e0 * 64, sAw, lane, 256);

    // Scatter-add new edge features into recvF (L2-resident, f32 atomics,
    // lanes cover contiguous 128B per row half).
    for (int r = 0; r < 32; ++r) {
        const int c = ei[E + e0 + r];
        float v0 = (float)sAw[r * 64 + lane];
        float v1 = (float)sAw[r * 64 + 32 + lane];
        atomic_add_f32(&recvF[(size_t)c * 64 + lane], v0);
        atomic_add_f32(&recvF[(size_t)c * 64 + 32 + lane], v1);
    }
}

// ---------------------------------------------------------------------------
// Node update: A = [xx | enc_x | bf16(recvF)] (16x192) -> 2-layer MLP -> xx
// ---------------------------------------------------------------------------
__global__ __launch_bounds__(128) void node_update_kernel(
    __bf16* __restrict__ xx, const __bf16* __restrict__ enc_x, const float* __restrict__ recvF,
    const unsigned* __restrict__ w1f, const float* __restrict__ b1,
    const unsigned* __restrict__ w2f, const float* __restrict__ b2, int ntiles) {
    __shared__ __bf16 sA[4][16 * 192];
    __shared__ __bf16 sH[4][16 * 64];
    const int wave = threadIdx.x >> 5, lane = threadIdx.x & 31;
    const int t = blockIdx.x * 4 + wave;
    if (t >= ntiles) return;
    const int n0 = t * 16;

    for (int r = 0; r < 16; ++r) {
        const size_t node = (size_t)(n0 + r);
        if (lane < 8) {
            ((uint4*)(sA[wave] + r * 192))[lane] = ((const uint4*)(xx + node * 64))[lane];
        } else if (lane < 16) {
            ((uint4*)(sA[wave] + r * 192))[lane] = ((const uint4*)(enc_x + node * 64))[lane & 7];
        } else {
            const int qq = lane - 16;  // 4 floats each
            float4 f = ((const float4*)(recvF + node * 64))[qq];
            __bf16* d = sA[wave] + r * 192 + 128 + qq * 4;
            d[0] = (__bf16)f.x; d[1] = (__bf16)f.y; d[2] = (__bf16)f.z; d[3] = (__bf16)f.w;
        }
    }

    v8f acc[4];
    zero_acc(acc);
    gemm_tile(sA[wave], 192, 6, w1f, lane, acc);   // 192 -> 64
    store_c_relu(sH[wave], acc, b1, lane);

    zero_acc(acc);
    gemm_tile(sH[wave], 64, 2, w2f, lane, acc);    // 64 -> 64
    store_c_relu(sA[wave], acc, b2, lane);

    copy_tiles_out(xx + (size_t)n0 * 64, sA[wave], lane, 128);
}

// ---------------------------------------------------------------------------
// Decoder: xx -> relu -> relu -> dot(w3)+b3 -> out[N] (f32)
// ---------------------------------------------------------------------------
__global__ __launch_bounds__(128) void decoder_kernel(
    const __bf16* __restrict__ xx,
    const unsigned* __restrict__ w1f, const float* __restrict__ b1,
    const unsigned* __restrict__ w2f, const float* __restrict__ b2,
    const float* __restrict__ w3, const float* __restrict__ b3,
    float* __restrict__ out, int ntiles) {
    __shared__ __bf16 sA[4][1024];
    __shared__ __bf16 sH[4][1024];
    const int wave = threadIdx.x >> 5, lane = threadIdx.x & 31;
    const int t = blockIdx.x * 4 + wave;
    if (t >= ntiles) return;
    const int n0 = t * 16;

    {
        const uint4* s = (const uint4*)(xx + (size_t)n0 * 64);
        uint4* d = (uint4*)sA[wave];
#pragma unroll
        for (int i = lane; i < 128; i += 32) d[i] = s[i];
    }

    v8f acc[4];
    zero_acc(acc);
    gemm_tile(sA[wave], 64, 2, w1f, lane, acc);
    store_c_relu(sH[wave], acc, b1, lane);

    zero_acc(acc);
    gemm_tile(sH[wave], 64, 2, w2f, lane, acc);
    store_c_relu(sA[wave], acc, b2, lane);

    if (lane < 16) {
        float s = 0.0f;
        for (int c = 0; c < 64; ++c) s += (float)sA[wave][lane * 64 + c] * w3[c];
        out[n0 + lane] = s + b3[0];
    }
}

// ---------------------------------------------------------------------------
// Host-side orchestration
// ---------------------------------------------------------------------------
extern "C" void kernel_launch(void* const* d_in, const int* in_sizes, int n_in,
                              void* d_out, int out_size, void* d_ws, size_t ws_size,
                              hipStream_t stream) {
    (void)n_in; (void)out_size; (void)ws_size;
    const float* x  = (const float*)d_in[0];
    const float* ea = (const float*)d_in[1];
    const int*   ei = (const int*)d_in[2];
    const float* ne_w1 = (const float*)d_in[3];
    const float* ne_b1 = (const float*)d_in[4];
    const float* ne_w2 = (const float*)d_in[5];
    const float* ne_b2 = (const float*)d_in[6];
    const float* ee_w1 = (const float*)d_in[7];
    const float* ee_b1 = (const float*)d_in[8];
    const float* ee_w2 = (const float*)d_in[9];
    const float* ee_b2 = (const float*)d_in[10];
    const float* ed_w1 = (const float*)d_in[11];
    const float* ed_b1 = (const float*)d_in[12];
    const float* ed_w2 = (const float*)d_in[13];
    const float* ed_b2 = (const float*)d_in[14];
    const float* no_w1 = (const float*)d_in[15];
    const float* no_b1 = (const float*)d_in[16];
    const float* no_w2 = (const float*)d_in[17];
    const float* no_b2 = (const float*)d_in[18];
    const float* de_w1 = (const float*)d_in[19];
    const float* de_b1 = (const float*)d_in[20];
    const float* de_w2 = (const float*)d_in[21];
    const float* de_b2 = (const float*)d_in[22];
    const float* de_w3 = (const float*)d_in[23];
    const float* de_b3 = (const float*)d_in[24];

    const int N = in_sizes[0];
    const int E = in_sizes[2] / 2;

    char* p = (char*)d_ws;
    auto carve = [&](size_t bytes) {
        char* r = p;
        p += (bytes + 255) & ~(size_t)255;
        return (void*)r;
    };
    __bf16* xx    = (__bf16*)carve((size_t)N * 64 * 2);
    __bf16* encx  = (__bf16*)carve((size_t)N * 64 * 2);
    __bf16* eef   = (__bf16*)carve((size_t)E * 64 * 2);
    __bf16* ence  = (__bf16*)carve((size_t)E * 64 * 2);
    float*  recvF = (float*)carve((size_t)N * 64 * 4);
    unsigned* f_ne_w2 = (unsigned*)carve(2048 * 4);
    unsigned* f_ee_w2 = (unsigned*)carve(2048 * 4);
    unsigned* f_ed_w1 = (unsigned*)carve(8192 * 4);
    unsigned* f_ed_w2 = (unsigned*)carve(2048 * 4);
    unsigned* f_no_w1 = (unsigned*)carve(6144 * 4);
    unsigned* f_no_w2 = (unsigned*)carve(2048 * 4);
    unsigned* f_de_w1 = (unsigned*)carve(2048 * 4);
    unsigned* f_de_w2 = (unsigned*)carve(2048 * 4);

    auto prep = [&](const float* W, unsigned* out_f, int K, int KT) {
        int total = KT * 4 * 256;
        prep_weight_kernel<<<(total + 255) / 256, 256, 0, stream>>>(W, out_f, K, KT, 4);
    };
    prep(ne_w2, f_ne_w2, 64, 2);
    prep(ee_w2, f_ee_w2, 64, 2);
    prep(ed_w1, f_ed_w1, 256, 8);
    prep(ed_w2, f_ed_w2, 64, 2);
    prep(no_w1, f_no_w1, 192, 6);
    prep(no_w2, f_no_w2, 64, 2);
    prep(de_w1, f_de_w1, 64, 2);
    prep(de_w2, f_de_w2, 64, 2);

    const int ntN = N / 16;
    const int ntE32 = E / 32;
    const int gN = (ntN + 3) / 4;
    const int gE = (ntE32 + 1) / 2;

    node_encoder_kernel<<<gN, 128, 0, stream>>>(x, ne_w1, ne_b1, f_ne_w2, ne_b2, encx, xx, ntN);
    edge_encoder_kernel<<<(E / 16 + 3) / 4, 128, 0, stream>>>(ea, ee_w1, ee_b1, f_ee_w2, ee_b2,
                                                              ence, eef, E / 16);

    for (int s = 0; s < 7; ++s) {
        zero_f32_kernel<<<((N * 64) + 255) / 256, 256, 0, stream>>>(recvF, N * 64);
        edge_update_kernel<<<gE, 64, 0, stream>>>(xx, ence, eef, ei, E,
                                                  f_ed_w1, ed_b1, f_ed_w2, ed_b2, recvF, ntE32);
        node_update_kernel<<<gN, 128, 0, stream>>>(xx, encx, recvF,
                                                   f_no_w1, no_b1, f_no_w2, no_b2, ntN);
    }

    decoder_kernel<<<gN, 128, 0, stream>>>(xx, f_de_w1, de_b1, f_de_w2, de_b2,
                                           de_w3, de_b3, (float*)d_out, ntN);
}
